// RNN_3702261809361
// MI455X (gfx1250) — compile-verified, matrix operand order checked by reading, not run on hardware
//
#include <hip/hip_runtime.h>
#include <hip/hip_bf16.h>

#define T_STEPS 256
#define BATCH   128
#define INPUT   1024
#define LATENT  2048
#define TK      32          // K-tile staged per iteration (bytes per row: 64)

typedef __attribute__((ext_vector_type(16))) __bf16 v16bf;
typedef __attribute__((ext_vector_type(8)))  float  v8f;

union Frag {
    uint4  u[2];
    v16bf  v;
};

// Per CDNA5 ISA A/B 16-bit layout: this lane holds K = {kb..kb+7, kb+16..kb+23}
// of its row (kb already includes the +8 shift for lanes 16..31).
__device__ __forceinline__ v16bf load_frag(const __hip_bfloat16* row, int kb) {
    Frag f;
    f.u[0] = *(const uint4*)(row + kb);        // K = kb .. kb+7
    f.u[1] = *(const uint4*)(row + kb + 16);   // K = kb+16 .. kb+23
    return f.v;
}

__device__ __forceinline__ v8f wmma_bf16(v16bf a, v16bf b, v8f c) {
    return __builtin_amdgcn_wmma_f32_16x16x32_bf16(false, a, false, b, (short)0, c,
                                                   false, false);
}

__device__ __forceinline__ float fast_tanh(float x) {
#if __has_builtin(__builtin_amdgcn_tanhf)
    return __builtin_amdgcn_tanhf(x);          // v_tanh_f32 (CDNA5 trans op)
#else
    float e = __expf(2.0f * x);                // v_exp_f32
    return 1.0f - 2.0f * __builtin_amdgcn_rcpf(e + 1.0f);
#endif
}

// Flat address of a __shared__ object keeps the LDS byte offset in addr[31:0]
// (ISA §10.2 aperture table), which is what the async-LDS VDST VGPR needs.
__device__ __forceinline__ unsigned lds_off(const void* p) {
    return (unsigned)(unsigned long long)p;
}

// CDNA5 async copy: 16 bytes global -> LDS, tracked by ASYNCcnt.
__device__ __forceinline__ void async16(unsigned lds_byte_addr, const void* gaddr) {
    unsigned long long a = (unsigned long long)gaddr;
    asm volatile("global_load_async_to_lds_b128 %0, %1, off scope:SCOPE_DEV"
                 :: "v"(lds_byte_addr), "v"(a)
                 : "memory");
}

__device__ __forceinline__ void wait_async_all() {
    asm volatile("s_wait_asynccnt 0x0" ::: "memory");
}

// Stage one 128x32 bf16 A tile and one 128x32 bf16 B tile (8 KB each) into LDS.
// 256 threads x 32 B per tile; rows are 64 B (32 bf16) in LDS.
__device__ __forceinline__ void stage_tiles(unsigned ldsA, unsigned ldsB,
                                            const __hip_bfloat16* __restrict__ Ag,
                                            const __hip_bfloat16* __restrict__ Bg,
                                            int ldk, int k0) {
    const int c   = threadIdx.x * 32;   // byte offset within the 8 KB tile
    const int row = c >> 6;             // 64 B per row
    const int off = c & 63;
    const char* ga = (const char*)(Ag + (size_t)row * ldk + k0) + off;
    const char* gb = (const char*)(Bg + (size_t)row * ldk + k0) + off;
    async16(ldsA + c,      ga);
    async16(ldsA + c + 16, ga + 16);
    async16(ldsB + c,      gb);
    async16(ldsB + c + 16, gb + 16);
}

// One 32-deep K step of the per-wave 32x64 tile: 2 A frags, 4 B frags, 8 WMMAs.
__device__ __forceinline__ void mma_step(const __hip_bfloat16* A, const __hip_bfloat16* B,
                                         int m0, int nn0, int sel, int ln,
                                         v8f acc[2][4]) {
    const int kb = sel * 8;
    v16bf a0 = load_frag(A + (m0 + ln) * TK, kb);
    v16bf a1 = load_frag(A + (m0 + 16 + ln) * TK, kb);
#pragma unroll
    for (int j = 0; j < 4; ++j) {
        v16bf b = load_frag(B + (nn0 + j * 16 + ln) * TK, kb);
        acc[0][j] = wmma_bf16(a0, b, acc[0][j]);
        acc[1][j] = wmma_bf16(a1, b, acc[1][j]);
    }
}

// Block-level 128x128 GEMM over K=KTOT with double-buffered async-LDS staging.
// Ag: 128 rows, pitch KTOT (bf16). Bg: 128 rows (= output columns), pitch KTOT.
template <int KTOT>
__device__ __forceinline__ void gemm128(const __hip_bfloat16* __restrict__ Ag,
                                        const __hip_bfloat16* __restrict__ Bg,
                                        __hip_bfloat16 (*As)[128 * TK],
                                        __hip_bfloat16 (*Bs)[128 * TK],
                                        int m0, int nn0, int sel, int ln,
                                        v8f acc[2][4]) {
    stage_tiles(lds_off(As[0]), lds_off(Bs[0]), Ag, Bg, KTOT, 0);
    int buf = 0;
    for (int k0 = 0; k0 < KTOT; k0 += TK) {
        wait_async_all();        // my prefetch into As/Bs[buf] complete
        __syncthreads();         // everyone's prefetch visible; prior reads done
        if (k0 + TK < KTOT)
            stage_tiles(lds_off(As[buf ^ 1]), lds_off(Bs[buf ^ 1]),
                        Ag, Bg, KTOT, k0 + TK);
        mma_step(As[buf], Bs[buf], m0, nn0, sel, ln, acc);
        buf ^= 1;
    }
}

// ---------------------------------------------------------------- converts ---
__global__ void cvt_bf16(const float* __restrict__ in, __hip_bfloat16* __restrict__ out,
                         long n) {
    long i = (long)blockIdx.x * blockDim.x + threadIdx.x;
    long stride = (long)gridDim.x * blockDim.x;
    for (; i < n; i += stride) out[i] = __float2bfloat16(in[i]);
}

// ----------------------------------------------------- xi = x @ Wi^T + b_i ---
__global__ void __launch_bounds__(256)
xi_gemm(const __hip_bfloat16* __restrict__ X, const __hip_bfloat16* __restrict__ Wi,
        const float* __restrict__ bi, float* __restrict__ xi) {
    __shared__ __hip_bfloat16 As[2][128 * TK];
    __shared__ __hip_bfloat16 Bs[2][128 * TK];

    const int lane = threadIdx.x & 31;
    const int wave = threadIdx.x >> 5;
    const int sel  = lane >> 4;
    const int ln   = lane & 15;
    const int mblk = blockIdx.y * 128;
    const int nblk = blockIdx.x * 128;
    const int m0   = (wave & 3) * 32;   // within-block
    const int nn0  = (wave >> 2) * 64;  // within-block

    v8f acc[2][4];
#pragma unroll
    for (int i = 0; i < 2; ++i)
#pragma unroll
        for (int j = 0; j < 4; ++j) acc[i][j] = (v8f){0,0,0,0,0,0,0,0};

    gemm128<INPUT>(X + (size_t)mblk * INPUT, Wi + (size_t)nblk * INPUT,
                   As, Bs, m0, nn0, sel, ln, acc);

#pragma unroll
    for (int i = 0; i < 2; ++i)
#pragma unroll
        for (int j = 0; j < 4; ++j) {
            const int n = nblk + nn0 + j * 16 + ln;
            const float bias = bi[n];
#pragma unroll
            for (int r = 0; r < 8; ++r) {
                const int m = mblk + m0 + i * 16 + sel * 8 + r;
                xi[(size_t)m * LATENT + n] = acc[i][j][r] + bias;
            }
        }
}

// ------------------------------------------------------------ persistent scan
__global__ void __launch_bounds__(256)
rnn_scan(const float* __restrict__ xi, const __hip_bfloat16* __restrict__ Wh,
         const float* __restrict__ bh, float* __restrict__ out,
         __hip_bfloat16* __restrict__ hbuf, unsigned* __restrict__ counter) {
    __shared__ __hip_bfloat16 As[2][128 * TK];
    __shared__ __hip_bfloat16 Bs[2][128 * TK];

    const int lane = threadIdx.x & 31;
    const int wave = threadIdx.x >> 5;
    const int sel  = lane >> 4;
    const int ln   = lane & 15;
    const int nblk = blockIdx.x * 128;
    const int m0   = (wave & 3) * 32;
    const int nn0  = (wave >> 2) * 64;
    const unsigned nblocks = gridDim.x;

    // ---- t = 0 : h0 = tanh(xi[0]) (no recurrent term)
    for (int idx = threadIdx.x; idx < 128 * 128; idx += 256) {
        const int m = idx >> 7;
        const int n = nblk + (idx & 127);
        const float v = fast_tanh(xi[(size_t)m * LATENT + n]);
        out[(size_t)m * LATENT + n]  = v;
        hbuf[(size_t)m * LATENT + n] = __float2bfloat16(v);  // buffer 0
    }

    const __hip_bfloat16* WhBlk = Wh + (size_t)nblk * LATENT;  // our 128 cols

    for (int t = 1; t < T_STEPS; ++t) {
        // ---- grid barrier: step t-1 h writes must be visible everywhere
        __threadfence();
        __syncthreads();
        if (threadIdx.x == 0) {
            atomicAdd(counter, 1u);
            const unsigned target = nblocks * (unsigned)t;
            while (__hip_atomic_load(counter, __ATOMIC_RELAXED,
                                     __HIP_MEMORY_SCOPE_AGENT) < target)
                __builtin_amdgcn_s_sleep(1);
        }
        __syncthreads();
        __threadfence();

        const __hip_bfloat16* H  = hbuf + (size_t)((t - 1) & 1) * (BATCH * LATENT);
        __hip_bfloat16*       Hn = hbuf + (size_t)(t & 1) * (BATCH * LATENT);
        const float* xit  = xi  + (size_t)t * BATCH * LATENT;
        float*       outt = out + (size_t)t * BATCH * LATENT;

        v8f acc[2][4];
#pragma unroll
        for (int i = 0; i < 2; ++i)
#pragma unroll
            for (int j = 0; j < 4; ++j) acc[i][j] = (v8f){0,0,0,0,0,0,0,0};

        gemm128<LATENT>(H, WhBlk, As, Bs, m0, nn0, sel, ln, acc);

        // fused epilogue: + xi_t + b_h, tanh, write f32 out + bf16 h
#pragma unroll
        for (int i = 0; i < 2; ++i)
#pragma unroll
            for (int j = 0; j < 4; ++j) {
                const int n = nblk + nn0 + j * 16 + ln;
                const float bias = bh[n];
#pragma unroll
                for (int r = 0; r < 8; ++r) {
                    const int m = m0 + i * 16 + sel * 8 + r;
                    const float v =
                        fast_tanh(acc[i][j][r] + xit[(size_t)m * LATENT + n] + bias);
                    outt[(size_t)m * LATENT + n] = v;
                    Hn[(size_t)m * LATENT + n]   = __float2bfloat16(v);
                }
            }
    }
}

// ----------------------------------------------------------------- launcher --
extern "C" void kernel_launch(void* const* d_in, const int* in_sizes, int n_in,
                              void* d_out, int out_size, void* d_ws, size_t ws_size,
                              hipStream_t stream) {
    (void)in_sizes; (void)n_in; (void)out_size; (void)ws_size;
    const float* x   = (const float*)d_in[0];   // [T,B,INPUT]
    const float* W_i = (const float*)d_in[1];   // [LATENT,INPUT]
    const float* b_i = (const float*)d_in[2];   // [LATENT]
    const float* W_h = (const float*)d_in[3];   // [LATENT,LATENT]
    const float* b_h = (const float*)d_in[4];   // [LATENT]
    float* out = (float*)d_out;                 // [T,B,LATENT]

    char* ws = (char*)d_ws;
    size_t off = 0;
    float* xi = (float*)(ws + off);
    off += (size_t)T_STEPS * BATCH * LATENT * sizeof(float);          // 268 MB
    __hip_bfloat16* xbf = (__hip_bfloat16*)(ws + off);
    off += (size_t)T_STEPS * BATCH * INPUT * sizeof(__hip_bfloat16);  // 67 MB
    __hip_bfloat16* wibf = (__hip_bfloat16*)(ws + off);
    off += (size_t)LATENT * INPUT * sizeof(__hip_bfloat16);           // 4 MB
    __hip_bfloat16* whbf = (__hip_bfloat16*)(ws + off);
    off += (size_t)LATENT * LATENT * sizeof(__hip_bfloat16);          // 8 MB
    __hip_bfloat16* hbuf = (__hip_bfloat16*)(ws + off);
    off += (size_t)2 * BATCH * LATENT * sizeof(__hip_bfloat16);       // 1 MB
    unsigned* counter = (unsigned*)(ws + off);
    off += 64;

    const long n_x  = (long)T_STEPS * BATCH * INPUT;
    const long n_wi = (long)LATENT * INPUT;
    const long n_wh = (long)LATENT * LATENT;
    cvt_bf16<<<4096, 256, 0, stream>>>(x,   xbf,  n_x);
    cvt_bf16<<<2048, 256, 0, stream>>>(W_i, wibf, n_wi);
    cvt_bf16<<<4096, 256, 0, stream>>>(W_h, whbf, n_wh);

    dim3 g1(LATENT / 128, (T_STEPS * BATCH) / 128);
    xi_gemm<<<g1, 256, 0, stream>>>(xbf, wibf, b_i, xi);

    hipMemsetAsync(counter, 0, 64, stream);   // graph-capturable memset node

    rnn_scan<<<LATENT / 128, 256, 0, stream>>>(xi, whbf, b_h, out, hbuf, counter);
}
